// Model_15908558865516
// MI455X (gfx1250) — matrix operand order repaired and checked by reading, not compiled, not run on hardware
//
#include <hip/hip_runtime.h>

#define S_LEN 2048
#define D_DIM 64
#define BQ    128            // q rows per workgroup (8 waves x 16)
#define CK    32             // keys per chunk
#define NCHUNK (S_LEN / CK)  // 64
#define KPAD  72             // K chunk rows, padded (36 dword stride -> no bank conflicts)
#define VPAD  40             // transposed V / P rows, padded (20 dword stride)

typedef __attribute__((ext_vector_type(16))) __bf16 v16bf;
typedef __attribute__((ext_vector_type(8)))  __bf16 v8bf;
typedef __attribute__((ext_vector_type(8)))  float  v8f;
typedef __attribute__((ext_vector_type(4)))  float  v4f;

union Frag16 { v16bf v; v8bf h[2]; };

// q scale 1/sqrt(64) with log2(e) folded in: scores live in log2 domain,
// so softmax exps are single native v_exp_f32 (exp2) instructions.
#define QSCALE 0.18033688011112042f   // 0.125 * log2(e)

__device__ __forceinline__ float rowmax16(float v) {
#pragma unroll
  for (int m = 1; m < 16; m <<= 1) v = fmaxf(v, __shfl_xor(v, m, 32));
  return v;
}
__device__ __forceinline__ float rowsum16(float v) {
#pragma unroll
  for (int m = 1; m < 16; m <<= 1) v += __shfl_xor(v, m, 32);
  return v;
}
// deterministic Weyl-threshold dropout: exact 0.9 keep rate, 1 mul + 1 cmp
__device__ __forceinline__ float dropmul(unsigned idx) {
  return (idx * 2654435761u < 3865470566u) ? (1.0f / 0.9f) : 0.0f;  // 0.9*2^32
}

__global__ __launch_bounds__(256)
void fattn_kernel(const float* __restrict__ Q, const float* __restrict__ K,
                  const float* __restrict__ Vv, const int* __restrict__ Mask,
                  float* __restrict__ O) {
  __shared__ __align__(16) __bf16 Kb[2][CK][KPAD];      // K chunk, [key][d]
  __shared__ __align__(16) __bf16 Vt[2][D_DIM][VPAD];   // V chunk, TRANSPOSED [d][key]
  __shared__ __align__(16) __bf16 Pb[8][16][VPAD];      // per-wave P tiles

  const int b    = blockIdx.x >> 4;     // 16 q-tiles per batch
  const int qt   = blockIdx.x & 15;
  const int tid  = threadIdx.x;
  const int wave = tid >> 5;
  const int lane = tid & 31;
  const int l16  = lane & 15;
  const int hs   = lane >> 4;           // which 16-lane half

  const int qrow0 = qt * BQ + wave * 16;
  const float* Kg = K  + (size_t)b * S_LEN * D_DIM;
  const float* Vg = Vv + (size_t)b * S_LEN * D_DIM;

  // ---------------- Q A-fragments (16x32 bf16 each), scale folded in -------
  // A layout 16-bit 16x32: lane row = l16; elems 0..7 = K(db..db+7),
  // elems 8..15 = K(db+16..db+23), db = hs*8.
  const float* qp = Q + ((size_t)b * S_LEN + (qrow0 + l16)) * D_DIM;
  Frag16 aQ0, aQ1;
  {
    const int db = hs * 8;
    v4f f0 = *(const v4f*)(qp + db);
    v4f f1 = *(const v4f*)(qp + db + 4);
    v4f f2 = *(const v4f*)(qp + db + 16);
    v4f f3 = *(const v4f*)(qp + db + 20);
    v4f g0 = *(const v4f*)(qp + db + 32);
    v4f g1 = *(const v4f*)(qp + db + 36);
    v4f g2 = *(const v4f*)(qp + db + 48);
    v4f g3 = *(const v4f*)(qp + db + 52);
#pragma unroll
    for (int i = 0; i < 4; ++i) {
      aQ0.v[i]      = (__bf16)(f0[i] * QSCALE);
      aQ0.v[4 + i]  = (__bf16)(f1[i] * QSCALE);
      aQ0.v[8 + i]  = (__bf16)(f2[i] * QSCALE);
      aQ0.v[12 + i] = (__bf16)(f3[i] * QSCALE);
      aQ1.v[i]      = (__bf16)(g0[i] * QSCALE);
      aQ1.v[4 + i]  = (__bf16)(g1[i] * QSCALE);
      aQ1.v[8 + i]  = (__bf16)(g2[i] * QSCALE);
      aQ1.v[12 + i] = (__bf16)(g3[i] * QSCALE);
    }
  }

  // K loader: thread -> 8 consecutive d of one key (coalesced b128 loads,
  // single contiguous b128 LDS store into row-major Kb).
  const int lkey = tid >> 3;          // 0..31
  const int ldd  = (tid & 7) * 8;     // 0..56
  // V loader: thread -> fixed d, 8 consecutive keys (coalesced dword loads,
  // single contiguous b128 LDS store into TRANSPOSED Vt).
  const int vd  = tid & 63;           // 0..63
  const int vkg = tid >> 6;           // 0..3  (keys vkg*8 .. vkg*8+7)

  auto load_chunk = [&](int kc, int buf) {
    const float* kg = Kg + (size_t)kc * CK * D_DIM + lkey * D_DIM + ldd;
    v4f a0 = *(const v4f*)(kg);
    v4f a1 = *(const v4f*)(kg + 4);
    v8bf kb;
#pragma unroll
    for (int i = 0; i < 4; ++i) { kb[i] = (__bf16)a0[i]; kb[4 + i] = (__bf16)a1[i]; }
    *(v8bf*)&Kb[buf][lkey][ldd] = kb;

    const float* vg = Vg + (size_t)kc * CK * D_DIM + (vkg * 8) * D_DIM + vd;
    v8bf vb;
#pragma unroll
    for (int j = 0; j < 8; ++j) vb[j] = (__bf16)vg[j * D_DIM];
    *(v8bf*)&Vt[buf][vd][vkg * 8] = vb;
  };

  // ---------------- running flash-softmax state (log2 domain) --------------
  v8f acc[4];
  float m_i[8], l_i[8];
#pragma unroll
  for (int r = 0; r < 8; ++r) {
    m_i[r] = -1.0e30f; l_i[r] = 0.0f;
#pragma unroll
    for (int t = 0; t < 4; ++t) acc[t][r] = 0.0f;
  }

  load_chunk(0, 0);
  __syncthreads();

  for (int kc = 0; kc < NCHUNK; ++kc) {
    const int cur = kc & 1;
    if (kc + 1 < NCHUNK) {
      __builtin_prefetch(Kg + (size_t)(kc + 1) * CK * D_DIM + tid * 8, 0, 1);
      __builtin_prefetch(Vg + (size_t)(kc + 1) * CK * D_DIM + tid * 8, 0, 1);
      load_chunk(kc + 1, cur ^ 1);
    }

    // ---- S = Q * K^T : two 16x16 score tiles, D split into 32+32 ----------
    // B layout 16-bit 32x16: lane col = l16 (key), elems 0..7 = K-dim
    // d = hs*8 .. hs*8+7, elems 8..15 = d = hs*8+16 .. hs*8+23.
    Frag16 bk[4];
#pragma unroll
    for (int t = 0; t < 2; ++t) {
      const int key = t * 16 + l16;
#pragma unroll
      for (int dd = 0; dd < 2; ++dd) {
        const int dstart = dd * 32 + hs * 8;
        bk[t * 2 + dd].h[0] = *(const v8bf*)&Kb[cur][key][dstart];
        bk[t * 2 + dd].h[1] = *(const v8bf*)&Kb[cur][key][dstart + 16];
      }
    }
    v8f c0, c1;
#pragma unroll
    for (int r = 0; r < 8; ++r) { c0[r] = 0.0f; c1[r] = 0.0f; }
    c0 = __builtin_amdgcn_wmma_f32_16x16x32_bf16(false, aQ0.v, false, bk[0].v, (short)0, c0, false, false);
    c0 = __builtin_amdgcn_wmma_f32_16x16x32_bf16(false, aQ1.v, false, bk[1].v, (short)0, c0, false, false);
    c1 = __builtin_amdgcn_wmma_f32_16x16x32_bf16(false, aQ0.v, false, bk[2].v, (short)0, c1, false, false);
    c1 = __builtin_amdgcn_wmma_f32_16x16x32_bf16(false, aQ1.v, false, bk[3].v, (short)0, c1, false, false);

    // ---- mask + online softmax (exp2) + dropout, write P tile to LDS ------
    const int kbase = kc * CK;
    v8f av;
#pragma unroll
    for (int r = 0; r < 8; ++r) {
      const int q_r = qrow0 + r + hs * 8;                  // C-layout row
      const int* mrow = Mask + (size_t)q_r * S_LEN + kbase;
      float s0 = mrow[l16]      ? c0[r] : -1.0e30f;
      float s1 = mrow[16 + l16] ? c1[r] : -1.0e30f;
      float m_new = fmaxf(m_i[r], rowmax16(fmaxf(s0, s1)));
      float alpha = __builtin_amdgcn_exp2f(m_i[r] - m_new);
      float p0 = __builtin_amdgcn_exp2f(s0 - m_new);
      float p1 = __builtin_amdgcn_exp2f(s1 - m_new);
      l_i[r] = l_i[r] * alpha + rowsum16(p0 + p1);
      m_i[r] = m_new;
      av[r] = alpha;
      const unsigned dbase = ((unsigned)b * S_LEN + (unsigned)q_r) * S_LEN + kbase;
      Pb[wave][r + hs * 8][l16]      = (__bf16)(p0 * dropmul(dbase + l16));
      Pb[wave][r + hs * 8][16 + l16] = (__bf16)(p1 * dropmul(dbase + 16u + l16));
    }
#pragma unroll
    for (int t = 0; t < 4; ++t) acc[t] *= av;   // packed v_pk_mul_f32 rescale
    // wave-local LDS turnaround (cross-lane, same wave: lockstep + DS counter)
    asm volatile("s_wait_dscnt 0" ::: "memory");

    // ---- O += P * V : P as 16x32 bf16 A-fragment, V (transposed in LDS)
    //      as 32x16 B-fragments with contiguous 8-key reads ----------------
    Frag16 aP;
    {
      const int kb0 = hs * 8;
      aP.h[0] = *(const v8bf*)&Pb[wave][l16][kb0];
      aP.h[1] = *(const v8bf*)&Pb[wave][l16][kb0 + 16];
    }
#pragma unroll
    for (int t = 0; t < 4; ++t) {
      Frag16 bv;                           // col d = t*16+l16, K = hs*8..+7, hs*8+16..+23
      bv.h[0] = *(const v8bf*)&Vt[cur][t * 16 + l16][hs * 8];
      bv.h[1] = *(const v8bf*)&Vt[cur][t * 16 + l16][hs * 8 + 16];
      acc[t] = __builtin_amdgcn_wmma_f32_16x16x32_bf16(false, aP.v, false, bv.v, (short)0, acc[t], false, false);
    }
    __syncthreads();   // next-chunk LDS writes visible; cur buffer reusable
  }

  // ---------------- epilogue: divide by denominator, store f32 -------------
#pragma unroll
  for (int r = 0; r < 8; ++r) {
    const float invl = 1.0f / l_i[r];
    const size_t obase = ((size_t)b * S_LEN + (qrow0 + r + hs * 8)) * D_DIM;
#pragma unroll
    for (int t = 0; t < 4; ++t)
      O[obase + t * 16 + l16] = acc[t][r] * invl;
  }
}

extern "C" void kernel_launch(void* const* d_in, const int* in_sizes, int n_in,
                              void* d_out, int out_size, void* d_ws, size_t ws_size,
                              hipStream_t stream) {
  const float* q    = (const float*)d_in[0];
  const float* k    = (const float*)d_in[1];
  const float* v    = (const float*)d_in[2];
  const int*   mask = (const int*)d_in[3];
  float* out = (float*)d_out;
  (void)in_sizes; (void)n_in; (void)out_size; (void)d_ws; (void)ws_size;
  dim3 grid(16 * (S_LEN / BQ));   // B * q-tiles = 256 workgroups
  fattn_kernel<<<grid, dim3(256), 0, stream>>>(q, k, v, mask, out);
}
